// CausalSelfAttention_83382495084659
// MI455X (gfx1250) — matrix operand order repaired and checked by reading, not compiled, not run on hardware
//
#include <hip/hip_runtime.h>

// ---- problem constants (from reference) ----
#define NH   12          // heads
#define HD   64          // head dim
#define BB   8           // batch
#define TT   1024        // seq len
#define CC   768         // channels
#define PP   4           // prefix length
#define TPP  1056        // padded key count: 33 chunks of 32 covers P+T=1028
#define N3   2304        // 3*C
#define MR   8192        // B*T

typedef _Float16 f16;
typedef __attribute__((ext_vector_type(16))) _Float16 v16h;
typedef __attribute__((ext_vector_type(8)))  _Float16 v8h;
typedef __attribute__((ext_vector_type(8)))  float    v8f;

// ---------------------------------------------------------------------------
// WMMA helpers (CDNA5 wave32, 16x16x32 f16 -> f32)
// ---------------------------------------------------------------------------
static __device__ __forceinline__ v8f wmma_f16(v16h a, v16h b, v8f c) {
  return __builtin_amdgcn_wmma_f32_16x16x32_f16(
      /*neg_a=*/false, a, /*neg_b=*/false, b,
      /*c_mod=*/(short)0, c, /*reuse_a=*/false, /*reuse_b=*/false);
}

static __device__ __forceinline__ v16h cat8(v8h lo, v8h hi) {
  v16h r;
#pragma unroll
  for (int i = 0; i < 8; ++i) { r[i] = lo[i]; r[8 + i] = hi[i]; }
  return r;
}

// A-fragment: 16x32 f16 tile from row-major memory (row stride ld halves).
// Contiguous k order (same as load_bT): lane l -> row l&15; K = kb..kb+15, kb=(l>>4)*16.
static __device__ __forceinline__ v16h load_a16x32(const f16* __restrict__ base,
                                                   int ld, int lane) {
  int row = lane & 15;
  int kb  = (lane >> 4) * 16;
  const f16* p = base + (size_t)row * ld + kb;
  v8h lo = *(const v8h*)p;
  v8h hi = *(const v8h*)(p + 8);
  return cat8(lo, hi);
}

// B-fragment: 32x16 f16 tile, sourced from the row-major TRANSPOSE of B
// (i.e. rows are the N dimension). Contiguous k order (as load_a16x32): lane l -> col l&15,
// K = kb2..kb2+15, kb2=(l>>4)*16  => one contiguous 32-byte load.
static __device__ __forceinline__ v16h load_bT(const f16* __restrict__ baseT,
                                               int ld, int lane) {
  int col = lane & 15;
  int kb  = (lane >> 4) * 16;
  return *(const v16h*)(baseT + (size_t)col * ld + kb);
}

// ---------------------------------------------------------------------------
// convert / pack kernels
// ---------------------------------------------------------------------------
__global__ void k_f32tof16(const float* __restrict__ src, f16* __restrict__ dst,
                           int n) {
  int i = blockIdx.x * blockDim.x + threadIdx.x;
  if (i < n) dst[i] = (f16)src[i];
}

// w[K][N] (row-major f32) -> wT[N][K] (row-major f16)
__global__ void k_transposeT(const float* __restrict__ w, f16* __restrict__ wT,
                             int K, int N) {
  int i = blockIdx.x * blockDim.x + threadIdx.x;
  if (i >= K * N) return;
  int k = i / N, n = i % N;
  wT[(size_t)n * K + k] = (f16)w[i];
}

// kv_cvec [B][P][2C] -> Kbuf [B,H,TPP,HD] rows 0..P-1, VT [B,H,HD,TPP] cols 0..P-1
__global__ void k_prefix(const float* __restrict__ kv, f16* __restrict__ Kb,
                         f16* __restrict__ VT) {
  int i = blockIdx.x * blockDim.x + threadIdx.x;
  const int tot = BB * PP * 2 * CC;
  if (i >= tot) return;
  int c2 = i % (2 * CC);
  int p  = (i / (2 * CC)) % PP;
  int b  = i / (2 * CC * PP);
  f16 v = (f16)kv[i];
  if (c2 < CC) {
    int h = c2 / HD, d = c2 % HD;
    Kb[(((size_t)b * NH + h) * TPP + p) * HD + d] = v;
  } else {
    int c = c2 - CC;
    int h = c / HD, d = c % HD;
    VT[(((size_t)b * NH + h) * HD + d) * TPP + p] = v;
  }
}

// ---------------------------------------------------------------------------
// QKV projection: qkvT[n][m] = sum_k wT[n][k] * x[m][k]  (+bias in epilogue)
// Register-blocked 32(n) x 64(m) per wave: 2 A-frags x 4 B-frags -> 8 WMMA
// per k-step on 12 b128 loads (1.5 loads per WMMA).
// Scatters to Q [B,H,T,HD] (scaled 1/sqrt(HD)), K [B,H,TPP,HD], VT [B,H,HD,TPP]
// grid (128, 9), block 256 (8 waves)
// ---------------------------------------------------------------------------
static __device__ __forceinline__ void qkv_scatter(
    v8f acc, int nt0, int mt0, int lane, const float* __restrict__ bias,
    f16* __restrict__ Qb, f16* __restrict__ Kb, f16* __restrict__ VT) {
  int m  = mt0 + (lane & 15);
  int b  = m >> 10;                // m / T
  int t  = m & (TT - 1);           // m % T
  int nb = nt0 + (lane >> 4) * 8;  // 8 consecutive n; same region & head
  int reg = nb / CC;               // 0=Q 1=K 2=V
  int nl  = nb - reg * CC;
  int h   = nl / HD, d0 = nl % HD;
  size_t bh = (size_t)b * NH + h;
  if (reg == 0) {
    f16* p = Qb + (bh * TT + t) * HD + d0;
#pragma unroll
    for (int r = 0; r < 8; ++r)
      p[r] = (f16)((acc[r] + bias[nb + r]) * 0.125f);   // fold 1/sqrt(64)
  } else if (reg == 1) {
    f16* p = Kb + (bh * TPP + (PP + t)) * HD + d0;
#pragma unroll
    for (int r = 0; r < 8; ++r) p[r] = (f16)(acc[r] + bias[nb + r]);
  } else {
    f16* p = VT + (bh * HD + d0) * TPP + (PP + t);
#pragma unroll
    for (int r = 0; r < 8; ++r) p[r * TPP] = (f16)(acc[r] + bias[nb + r]);
  }
}

__global__ void __launch_bounds__(256)
k_qkv(const f16* __restrict__ wT, const f16* __restrict__ x16,
      const float* __restrict__ bias, f16* __restrict__ Qb,
      f16* __restrict__ Kb, f16* __restrict__ VT) {
  int lane = threadIdx.x & 31;
  int wv   = threadIdx.x >> 5;
  int m0   = blockIdx.x * 64;                 // 0..8191 step 64
  int n0   = (blockIdx.y * 8 + wv) * 32;      // 0..2303 step 32

  const f16* aB = wT + (size_t)n0 * CC;
  const f16* bB = x16 + (size_t)m0 * CC;
  v8f acc[2][4] = {};
#pragma unroll 2
  for (int k0 = 0; k0 < CC; k0 += 32) {
    __builtin_prefetch(aB + k0 + 256, 0, 1);              // global_prefetch_b8
    __builtin_prefetch(bB + k0 + 256, 0, 1);
    v16h a0 = load_a16x32(aB + k0, CC, lane);
    v16h a1 = load_a16x32(aB + (size_t)16 * CC + k0, CC, lane);
    v16h b0 = load_bT(bB + k0, CC, lane);
    v16h b1 = load_bT(bB + (size_t)16 * CC + k0, CC, lane);
    v16h b2 = load_bT(bB + (size_t)32 * CC + k0, CC, lane);
    v16h b3 = load_bT(bB + (size_t)48 * CC + k0, CC, lane);
    acc[0][0] = wmma_f16(a0, b0, acc[0][0]);
    acc[0][1] = wmma_f16(a0, b1, acc[0][1]);
    acc[0][2] = wmma_f16(a0, b2, acc[0][2]);
    acc[0][3] = wmma_f16(a0, b3, acc[0][3]);
    acc[1][0] = wmma_f16(a1, b0, acc[1][0]);
    acc[1][1] = wmma_f16(a1, b1, acc[1][1]);
    acc[1][2] = wmma_f16(a1, b2, acc[1][2]);
    acc[1][3] = wmma_f16(a1, b3, acc[1][3]);
  }
#pragma unroll
  for (int i = 0; i < 2; ++i)
#pragma unroll
    for (int j = 0; j < 4; ++j)
      qkv_scatter(acc[i][j], n0 + i * 16, m0 + j * 16, lane, bias, Qb, Kb, VT);
}

// ---------------------------------------------------------------------------
// Flash attention (transposed): one wave = one (b,h,qtile of 16 queries).
// S^T[key][q] = K·Q^T ;  O^T[d][q] = V^T·P^T ; softmax stats per lane (per q).
// grid 768, block 256 (8 waves)
// ---------------------------------------------------------------------------
__global__ void __launch_bounds__(256)
k_attn(const f16* __restrict__ Qb, const f16* __restrict__ Kb,
       const f16* __restrict__ VT, f16* __restrict__ Y) {
  int lane = threadIdx.x & 31;
  int wv   = threadIdx.x >> 5;
  int task = blockIdx.x * 8 + wv;   // 0..6143
  int qt   = task & 63;
  int bh   = task >> 6;             // b*NH + h
  int q0   = qt * 16;

  const f16* Qbase = Qb + ((size_t)bh * TT + q0) * HD;
  const f16* Kbase = Kb + (size_t)bh * TPP * HD;
  const f16* Vbase = VT + (size_t)bh * HD * TPP;

  // Q^T B-fragments for the two d-ksteps (d 0..31, 32..63): contiguous rows of Q
  v16h qf0 = load_bT(Qbase + 0, HD, lane);
  v16h qf1 = load_bT(Qbase + 32, HD, lane);

  v8f o[4] = {v8f{}, v8f{}, v8f{}, v8f{}};
  float m_run = -1e30f, l_run = 0.f;
  int q    = q0 + (lane & 15);          // this lane's query column
  int keyl = (lane >> 4) * 8;           // local key row base within score tile
  bool hiHalf = lane >= 16;

  int kmax = PP + q0 + 16;              // last key needed (exclusive)
  int nch  = (kmax + 31) >> 5;          // 32-key chunks (max 33 -> TPP bound)
  for (int kc = 0; kc < nch; ++kc) {
    int kbase = kc * 32;
    const f16* K0 = Kbase + (size_t)kbase * HD;
    v8f s0 = {}, s1 = {};
    s0 = wmma_f16(load_a16x32(K0, HD, lane), qf0, s0);
    s0 = wmma_f16(load_a16x32(K0 + 32, HD, lane), qf1, s0);
    s1 = wmma_f16(load_a16x32(K0 + 16 * HD, HD, lane), qf0, s1);
    s1 = wmma_f16(load_a16x32(K0 + 16 * HD + 32, HD, lane), qf1, s1);

    // mask (prefix fully visible, causal beyond) + chunk max
    float mc = -1e30f;
#pragma unroll
    for (int r = 0; r < 8; ++r) {
      int k0i = kbase + keyl + r;
      int k1i = k0i + 16;
      s0[r] = ((k0i < PP) || (k0i - PP <= q)) ? s0[r] : -1e30f;
      s1[r] = ((k1i < PP) || (k1i - PP <= q)) ? s1[r] : -1e30f;
      mc = fmaxf(mc, fmaxf(s0[r], s1[r]));
    }
    mc = fmaxf(mc, __shfl_xor(mc, 16, 32));
    float m_new = fmaxf(m_run, mc);
    float scl   = __expf(m_run - m_new);

    float p0[8], p1[8], lc = 0.f;
#pragma unroll
    for (int r = 0; r < 8; ++r) {
      p0[r] = __expf(s0[r] - m_new);
      p1[r] = __expf(s1[r] - m_new);
      lc += p0[r] + p1[r];
    }
    lc += __shfl_xor(lc, 16, 32);
    l_run = l_run * scl + lc;
    m_run = m_new;
#pragma unroll
    for (int dt = 0; dt < 4; ++dt)
#pragma unroll
      for (int r = 0; r < 8; ++r) o[dt][r] *= scl;

    // build P^T B-fragment (lane=col q, K=keys) via cross-half shuffles
    v16h pb;
#pragma unroll
    for (int r = 0; r < 8; ++r) {
      float x0 = __shfl_xor(p0[r], 16, 32);
      float x1 = __shfl_xor(p1[r], 16, 32);
      pb[r]     = (f16)(hiHalf ? x1 : p0[r]);
      pb[8 + r] = (f16)(hiHalf ? p1[r] : x0);
    }

    // O^T += V^T(chunk) x P^T : 4 d-tiles of 16
#pragma unroll
    for (int dt = 0; dt < 4; ++dt) {
      const f16* Vp = Vbase + (size_t)(dt * 16) * TPP + kbase;
      o[dt] = wmma_f16(load_a16x32(Vp, TPP, lane), pb, o[dt]);
    }
  }

  float inv = 1.f / l_run;
  int dl0 = (lane >> 4) * 8;
  int b = bh / NH, h = bh % NH;
  f16* Yp = Y + ((size_t)b * TT + q) * CC + h * HD;
#pragma unroll
  for (int dt = 0; dt < 4; ++dt)
#pragma unroll
    for (int r = 0; r < 8; ++r)
      Yp[dt * 16 + dl0 + r] = (f16)(o[dt][r] * inv);
}

// ---------------------------------------------------------------------------
// Output projection: outT[n][m] = sum_k wprojT[n][k] * Y[m][k] + bias[n]
// Register-blocked 32(n) x 64(m) per wave. grid (128, 3), block 256
// ---------------------------------------------------------------------------
__global__ void __launch_bounds__(256)
k_proj(const f16* __restrict__ wT, const f16* __restrict__ Y,
       const float* __restrict__ bias, float* __restrict__ out) {
  int lane = threadIdx.x & 31;
  int wv   = threadIdx.x >> 5;
  int m0   = blockIdx.x * 64;
  int n0   = (blockIdx.y * 8 + wv) * 32;

  const f16* aB = wT + (size_t)n0 * CC;
  const f16* bB = Y + (size_t)m0 * CC;
  v8f acc[2][4] = {};
#pragma unroll 2
  for (int k0 = 0; k0 < CC; k0 += 32) {
    __builtin_prefetch(aB + k0 + 256, 0, 1);
    __builtin_prefetch(bB + k0 + 256, 0, 1);
    v16h a0 = load_a16x32(aB + k0, CC, lane);
    v16h a1 = load_a16x32(aB + (size_t)16 * CC + k0, CC, lane);
    v16h b0 = load_bT(bB + k0, CC, lane);
    v16h b1 = load_bT(bB + (size_t)16 * CC + k0, CC, lane);
    v16h b2 = load_bT(bB + (size_t)32 * CC + k0, CC, lane);
    v16h b3 = load_bT(bB + (size_t)48 * CC + k0, CC, lane);
    acc[0][0] = wmma_f16(a0, b0, acc[0][0]);
    acc[0][1] = wmma_f16(a0, b1, acc[0][1]);
    acc[0][2] = wmma_f16(a0, b2, acc[0][2]);
    acc[0][3] = wmma_f16(a0, b3, acc[0][3]);
    acc[1][0] = wmma_f16(a1, b0, acc[1][0]);
    acc[1][1] = wmma_f16(a1, b1, acc[1][1]);
    acc[1][2] = wmma_f16(a1, b2, acc[1][2]);
    acc[1][3] = wmma_f16(a1, b3, acc[1][3]);
  }
#pragma unroll
  for (int i = 0; i < 2; ++i)
#pragma unroll
    for (int j = 0; j < 4; ++j) {
      int m  = m0 + j * 16 + (lane & 15);
      int nb = n0 + i * 16 + (lane >> 4) * 8;
      float* p = out + (size_t)m * CC + nb;
#pragma unroll
      for (int r = 0; r < 8; ++r) p[r] = acc[i][j][r] + bias[nb + r];
    }
}

// ---------------------------------------------------------------------------
extern "C" void kernel_launch(void* const* d_in, const int* in_sizes, int n_in,
                              void* d_out, int out_size, void* d_ws,
                              size_t ws_size, hipStream_t stream) {
  (void)in_sizes; (void)n_in; (void)out_size; (void)ws_size;
  const float* x      = (const float*)d_in[0];   // [B,T,C]
  const float* kv     = (const float*)d_in[1];   // [B,P,2C]
  const float* w_attn = (const float*)d_in[2];   // [C,3C]
  const float* b_attn = (const float*)d_in[3];   // [3C]
  const float* w_proj = (const float*)d_in[4];   // [C,C]
  const float* b_proj = (const float*)d_in[5];   // [C]
  float* out = (float*)d_out;                    // [B,T,C]

  size_t off = 0;
  auto carve = [&](size_t bytes) -> char* {
    char* p = (char*)d_ws + off;
    off += (bytes + 255) & ~(size_t)255;
    return p;
  };
  f16* x16    = (f16*)carve((size_t)MR * CC * 2);
  f16* wattnT = (f16*)carve((size_t)N3 * CC * 2);
  f16* wprojT = (f16*)carve((size_t)CC * CC * 2);
  f16* Qb     = (f16*)carve((size_t)BB * NH * TT * HD * 2);
  f16* Kb     = (f16*)carve((size_t)BB * NH * TPP * HD * 2);
  f16* VTb    = (f16*)carve((size_t)BB * NH * HD * TPP * 2);
  f16* Y16    = (f16*)carve((size_t)MR * CC * 2);

  k_f32tof16<<<(MR * CC + 255) / 256, 256, 0, stream>>>(x, x16, MR * CC);
  k_transposeT<<<(CC * N3 + 255) / 256, 256, 0, stream>>>(w_attn, wattnT, CC, N3);
  k_transposeT<<<(CC * CC + 255) / 256, 256, 0, stream>>>(w_proj, wprojT, CC, CC);
  k_prefix<<<(BB * PP * 2 * CC + 255) / 256, 256, 0, stream>>>(kv, Kb, VTb);

  k_qkv<<<dim3(128, 9), 256, 0, stream>>>(wattnT, x16, b_attn, Qb, Kb, VTb);
  k_attn<<<768, 256, 0, stream>>>(Qb, Kb, VTb, Y16);
  k_proj<<<dim3(128, 3), 256, 0, stream>>>(wprojT, Y16, b_proj, out);
}
